// BaseSTGCNLayer_21500606284425
// MI455X (gfx1250) — compile-verified, matrix operand order checked by reading, not run on hardware
//
#include <hip/hip_runtime.h>

// ---------------------------------------------------------------------------
// BaseSTGCNLayer for MI455X (gfx1250, wave32, WMMA).
//   T=12, C_in=C_out=64, N nodes, E edges.
// Dense math (temporal conv as [16x192]x[192x64] GEMM and the GCN 64x64
// linear) runs on v_wmma_f32_16x16x32_f16 with branch-free, 16-byte-vector
// LDS fragment loads.  Edge aggregation is L2-resident (xw + out << 192MB L2)
// atomic scatter.
// ---------------------------------------------------------------------------

typedef __attribute__((ext_vector_type(16))) _Float16 v16h;
typedef __attribute__((ext_vector_type(8)))  _Float16 v8h;
typedef __attribute__((ext_vector_type(2)))  _Float16 v2h;
typedef __attribute__((ext_vector_type(8)))  float    v8f;

#define TDIM   12
#define CH     64
#define KCONV  192     // 3 taps * 64 in-channels
#define XROWS  18      // padded rows: t = -1 .. 16
#define XWPAD  1024    // padded xw row stride (16*64), t>=12 unused

__device__ __forceinline__ v8f vzero8() {
  v8f z;
#pragma unroll
  for (int i = 0; i < 8; ++i) z[i] = 0.0f;
  return z;
}

// 16-bit WMMA fragment: lane's 16 elements are two contiguous 8-half runs:
//   K = base+kb .. base+kb+7   and   K = base+kb+16 .. base+kb+23,
// where kb = (lane>>4)*8.  Load as two ds_load_b128 and concatenate.
__device__ __forceinline__ v16h load_frag16(const _Float16* p) {
  v8h lo = *(const v8h*)p;
  v8h hi = *(const v8h*)(p + 16);
  return __builtin_shufflevector(lo, hi, 0, 1, 2, 3, 4, 5, 6, 7,
                                 8, 9, 10, 11, 12, 13, 14, 15);
}

// ---------------------------------------------------------------------------
// Kernel 1: softmax stats over all edge weights (max, sum of exp).
// ---------------------------------------------------------------------------
__global__ __launch_bounds__(1024)
void softmax_reduce_kernel(const float* __restrict__ ew, int E,
                           float* __restrict__ red) {
  __shared__ float sm[1024];
  const int tid = threadIdx.x;

  float m = -3.402823466e+38f;
  for (int i = tid; i < E; i += 1024) m = fmaxf(m, ew[i]);
  sm[tid] = m;
  __syncthreads();
  for (int s = 512; s > 0; s >>= 1) {
    if (tid < s) sm[tid] = fmaxf(sm[tid], sm[tid + s]);
    __syncthreads();
  }
  const float mx = sm[0];
  __syncthreads();

  float acc = 0.0f;
  for (int i = tid; i < E; i += 1024) acc += __expf(ew[i] - mx);
  sm[tid] = acc;
  __syncthreads();
  for (int s = 512; s > 0; s >>= 1) {
    if (tid < s) sm[tid] += sm[tid + s];
    __syncthreads();
  }
  if (tid == 0) { red[0] = mx; red[1] = sm[0]; }
}

// ---------------------------------------------------------------------------
// Kernel 2a/2b/2c: degree with self-loops -> dis = rsqrt(deg)
// ---------------------------------------------------------------------------
__global__ void deg_init_kernel(float* __restrict__ deg, int N) {
  int n = blockIdx.x * blockDim.x + threadIdx.x;
  if (n < N) deg[n] = 1.0f;  // self-loop weight
}

__global__ void deg_accum_kernel(const int* __restrict__ ei,
                                 const float* __restrict__ ew,
                                 const float* __restrict__ red,
                                 float* __restrict__ deg, int E) {
  int e = blockIdx.x * blockDim.x + threadIdx.x;
  if (e < E) {
    float w = __expf(ew[e] - red[0]) / red[1];
    atomicAdd(&deg[ei[E + e]], w);  // col-indexed segment sum
  }
}

__global__ void deg_to_dis_kernel(float* __restrict__ deg, int N) {
  int n = blockIdx.x * blockDim.x + threadIdx.x;
  if (n < N) {
    float d = deg[n];
    deg[n] = (d > 0.0f) ? __frsqrt_rn(d) : 0.0f;
  }
}

// ---------------------------------------------------------------------------
// Kernel 3 (the WMMA kernel): per node:
//   conv1d(k=3,pad=1) as GEMM  ->  +bias, clip  ->  InstanceNorm over T
//   -> *gamma+beta -> relu -> GEMM with gcn_w^T  -> xw[n, t, o] (stride 1024)
// One wave (32 lanes) per node; 4 waves / 128-thread block.
// Branch-free: padded x tile (18 rows) makes all fragment loads in-bounds;
// garbage in D rows t>=12 is never consumed.
// ---------------------------------------------------------------------------
__global__ __launch_bounds__(128)
void fused_tconv_gcnlin_kernel(const float* __restrict__ x,
                               const float* __restrict__ conv_w,
                               const float* __restrict__ conv_b,
                               const float* __restrict__ gamma,
                               const float* __restrict__ beta,
                               const float* __restrict__ gcn_w,
                               float* __restrict__ xw, int N) {
  __shared__ __align__(16) _Float16 sBconv[CH * KCONV];  // [o][k], k=tap*64+cin
  __shared__ __align__(16) _Float16 sBgcn[CH * CH];      // [o][c] (native)
  __shared__ __align__(16) _Float16 sTile[4][XROWS * CH];

  const int tid  = threadIdx.x;
  const int lane = tid & 31;
  const int wave = tid >> 5;
  const int node = blockIdx.x * 4 + wave;
  const int col  = lane & 15;       // M row (A/D) and N col (B/D) for this lane
  const int half = lane >> 4;       // 0: D rows 0-7, 1: D rows 8-15
  const int kb   = half << 3;       // fragment K sub-offset

  // ---- stage conv weights o-major: sBconv[o*192 + tap*64 + cin] ----
  for (int idx = tid; idx < CH * KCONV; idx += 128) {
    int o = idx / KCONV;
    int k = idx - o * KCONV;
    int cin = k & 63, tap = k >> 6;
    sBconv[idx] = (_Float16)conv_w[o * KCONV + cin * 3 + tap];
  }
  // ---- stage gcn weights (already o-major: gcn_w[o][c]) ----
  for (int idx = tid; idx < CH * CH; idx += 128) {
    sBgcn[idx] = (_Float16)gcn_w[idx];
  }

  // ---- stage x[.,node,.] as fp16; rows: 0 = t=-1, 1..12 = t=0..11,
  //      13..17 = zero pad (keeps every fragment address in-bounds) ----
  _Float16* sX = &sTile[wave][0];
  {
    v2h z2;
    z2[0] = (_Float16)0.0f;
    z2[1] = (_Float16)0.0f;
    // zero rows 0 and 13..17  (6 rows * 32 v2h slots)
#pragma unroll
    for (int r = 0; r < 3; ++r) {
      *(v2h*)(sX + 0 * CH + ((lane + r * 32) & 31) * 2 + (r / 3) * 0) = z2;
    }
    // simpler deterministic zeroing: 6 rows, each 32 x v2h
    const int zrow[6] = {0, 13, 14, 15, 16, 17};
#pragma unroll
    for (int r = 0; r < 6; ++r) {
      *(v2h*)(sX + zrow[r] * CH + lane * 2) = z2;
    }
  }
  if (node < N) {
    const float2* xs =
        (const float2*)(x + (size_t)node * CH);  // + t*N*CH below
#pragma unroll
    for (int it = 0; it < TDIM; ++it) {
      // 32 lanes cover one row of 64 channels as float2
      float2 f = *(const float2*)(x + (size_t)it * N * CH +
                                  (size_t)node * CH + lane * 2);
      v2h p;
      p[0] = (_Float16)f.x;
      p[1] = (_Float16)f.y;
      *(v2h*)(sX + (it + 1) * CH + lane * 2) = p;
    }
    (void)xs;
  }
  __syncthreads();

  // ---- conv GEMM: [16 x 192] x [192 x 64], 6 K-steps x 4 N-tiles ----
  v8f acc[4];
#pragma unroll
  for (int nt = 0; nt < 4; ++nt) acc[nt] = vzero8();

#pragma unroll
  for (int ks = 0; ks < 6; ++ks) {
    const int tap = ks >> 1;            // 32-wide K block sits in one tap
    const int cb  = (ks & 1) * 32;      // cin base within tap
    v16h a = load_frag16(sX + (col + tap) * CH + cb + kb);
#pragma unroll
    for (int nt = 0; nt < 4; ++nt) {
      const int o = nt * 16 + col;
      v16h b = load_frag16(sBconv + o * KCONV + ks * 32 + kb);
      acc[nt] = __builtin_amdgcn_wmma_f32_16x16x32_f16(
          false, a, false, b, (short)0, acc[nt], false, false);
    }
  }

  // ---- per-channel params for this lane's 4 columns ----
  float bia[4], ga[4], be[4];
#pragma unroll
  for (int nt = 0; nt < 4; ++nt) {
    int o = nt * 16 + col;
    bia[nt] = conv_b[o];
    ga[nt]  = gamma[o];
    be[nt]  = beta[o];
  }

  // ---- InstanceNorm stats over T (M rows split across lane^16 pair) ----
  float s1[4], s2[4];
#pragma unroll
  for (int nt = 0; nt < 4; ++nt) { s1[nt] = 0.0f; s2[nt] = 0.0f; }
#pragma unroll
  for (int nt = 0; nt < 4; ++nt) {
#pragma unroll
    for (int r = 0; r < 8; ++r) {
      float v = acc[nt][r] + bia[nt];
      v = fminf(fmaxf(v, -10.0f), 10.0f);
      float use = (half == 0 || r < 4) ? 1.0f : 0.0f;  // valid t only
      s1[nt] += use * v;
      s2[nt] += use * v * v;
    }
  }
#pragma unroll
  for (int nt = 0; nt < 4; ++nt) {
    s1[nt] += __shfl_xor(s1[nt], 16, 32);
    s2[nt] += __shfl_xor(s2[nt], 16, 32);
  }

  __syncthreads();  // conv reads of sX complete; tile becomes h

  // ---- normalize, affine, relu -> h tile (fp16) rows 0..15 ----
  _Float16* sH = sX;  // reuse first 16 rows
#pragma unroll
  for (int nt = 0; nt < 4; ++nt) {
    float mean = s1[nt] * (1.0f / 12.0f);
    float var  = s2[nt] * (1.0f / 12.0f) - mean * mean;
    float rsig = __frsqrt_rn(var + 1e-5f);
#pragma unroll
    for (int r = 0; r < 8; ++r) {
      int t = half * 8 + r;
      float v = acc[nt][r] + bia[nt];
      v = fminf(fmaxf(v, -10.0f), 10.0f);
      v = (v - mean) * rsig * ga[nt] + be[nt];
      v = fmaxf(v, 0.0f);
      float hv = (t < TDIM) ? v : 0.0f;   // branch-free select, store always
      sH[t * CH + nt * 16 + col] = (_Float16)hv;
    }
  }
  __syncthreads();

  // ---- GCN linear GEMM: [16 x 64] x [64 x 64], 2 K-steps x 4 N-tiles ----
  v8f acc2[4];
#pragma unroll
  for (int nt = 0; nt < 4; ++nt) acc2[nt] = vzero8();

#pragma unroll
  for (int ks = 0; ks < 2; ++ks) {
    v16h a = load_frag16(sH + col * CH + ks * 32 + kb);
#pragma unroll
    for (int nt = 0; nt < 4; ++nt) {
      const int o = nt * 16 + col;
      v16h b = load_frag16(sBgcn + o * CH + ks * 32 + kb);
      acc2[nt] = __builtin_amdgcn_wmma_f32_16x16x32_f16(
          false, a, false, b, (short)0, acc2[nt], false, false);
    }
  }

  // ---- store xw[n][t][o], padded row stride 1024; unconditional ----
  if (node < N) {
    float* dst = xw + (size_t)node * XWPAD;
#pragma unroll
    for (int nt = 0; nt < 4; ++nt) {
#pragma unroll
      for (int r = 0; r < 8; ++r) {
        int t = half * 8 + r;                       // 0..15, pad rows unused
        dst[t * CH + nt * 16 + col] = acc2[nt][r];
      }
    }
  }
}

// ---------------------------------------------------------------------------
// Kernel 4: out = dis[n]^2 * xw[n]   (self loop; also initializes d_out)
// ---------------------------------------------------------------------------
__global__ void selfloop_init_kernel(const float* __restrict__ xw,
                                     const float* __restrict__ dis,
                                     float* __restrict__ out, int total) {
  int gid = blockIdx.x * blockDim.x + threadIdx.x;
  if (gid < total) {
    int n = gid / (TDIM * CH);
    int j = gid - n * (TDIM * CH);
    float d = dis[n];
    out[gid] = d * d * xw[(size_t)n * XWPAD + j];
  }
}

// ---------------------------------------------------------------------------
// Kernel 5: per-edge scatter-add:  out[col] += norm_e * xw[row]
// One 256-thread block per edge; 768 contiguous floats (L2-resident).
// ---------------------------------------------------------------------------
__global__ __launch_bounds__(256)
void edge_scatter_kernel(const int* __restrict__ ei,
                         const float* __restrict__ ew,
                         const float* __restrict__ red,
                         const float* __restrict__ dis,
                         const float* __restrict__ xw,
                         float* __restrict__ out, int E) {
  int e = blockIdx.x;
  int r = ei[e];
  int c = ei[E + e];
  float w    = __expf(ew[e] - red[0]) / red[1];
  float norm = dis[r] * w * dis[c];
  const float* src = xw  + (size_t)r * XWPAD;
  float*       dst = out + (size_t)c * (TDIM * CH);
  __builtin_prefetch(src + threadIdx.x, 0, 0);   // global_prefetch_b8
#pragma unroll
  for (int k = 0; k < 3; ++k) {
    int j = threadIdx.x + k * 256;
    atomicAdd(&dst[j], norm * src[j]);
  }
}

// ---------------------------------------------------------------------------
// Kernel 6: out = clip(relu(out + gcn_b), -10, 10)
// ---------------------------------------------------------------------------
__global__ void finalize_kernel(float* __restrict__ out,
                                const float* __restrict__ gcn_b, int total) {
  int gid = blockIdx.x * blockDim.x + threadIdx.x;
  if (gid < total) {
    float v = out[gid] + gcn_b[gid & (CH - 1)];
    out[gid] = fminf(fmaxf(v, 0.0f), 10.0f);
  }
}

// ---------------------------------------------------------------------------
extern "C" void kernel_launch(void* const* d_in, const int* in_sizes, int n_in,
                              void* d_out, int out_size, void* d_ws,
                              size_t ws_size, hipStream_t stream) {
  const float* x      = (const float*)d_in[0];
  const int*   ei     = (const int*)d_in[1];
  const float* ew     = (const float*)d_in[2];
  const float* conv_w = (const float*)d_in[3];
  const float* conv_b = (const float*)d_in[4];
  const float* gamma  = (const float*)d_in[5];
  const float* beta   = (const float*)d_in[6];
  const float* gcn_w  = (const float*)d_in[7];
  const float* gcn_b  = (const float*)d_in[8];

  const int E = in_sizes[2];
  const int N = in_sizes[0] / (TDIM * CH);
  const int total = N * TDIM * CH;

  float* wsf = (float*)d_ws;
  float* red = wsf;             // [0]=max, [1]=sum(exp)
  float* dis = wsf + 16;        // N floats: deg -> rsqrt(deg)
  float* xw  = wsf + 16 + N;    // N*1024 floats (padded rows)

  float* out = (float*)d_out;

  softmax_reduce_kernel<<<1, 1024, 0, stream>>>(ew, E, red);
  deg_init_kernel<<<(N + 255) / 256, 256, 0, stream>>>(dis, N);
  deg_accum_kernel<<<(E + 255) / 256, 256, 0, stream>>>(ei, ew, red, dis, E);
  deg_to_dis_kernel<<<(N + 255) / 256, 256, 0, stream>>>(dis, N);

  fused_tconv_gcnlin_kernel<<<(N + 3) / 4, 128, 0, stream>>>(
      x, conv_w, conv_b, gamma, beta, gcn_w, xw, N);

  selfloop_init_kernel<<<(total + 255) / 256, 256, 0, stream>>>(xw, dis, out,
                                                                total);
  edge_scatter_kernel<<<E, 256, 0, stream>>>(ei, ew, red, dis, xw, out, E);
  finalize_kernel<<<(total + 255) / 256, 256, 0, stream>>>(out, gcn_b, total);
}
